// TernaryLinear_14259291422934
// MI455X (gfx1250) — matrix operand order repaired and checked by reading, not compiled, not run on hardware
//
#include <hip/hip_runtime.h>

typedef __attribute__((ext_vector_type(16))) _Float16 v16h;
typedef __attribute__((ext_vector_type(8)))  _Float16 v8h;
typedef __attribute__((ext_vector_type(4)))  _Float16 v4h;
typedef __attribute__((ext_vector_type(8)))  float    v8f;

#define THRESH 0.75f
#define EPS_V  1e-6f

#define BM 128
#define BN 128
#define BK 32
#define LDT 40                 // padded LDS row stride in f16 (80 B, 16B-aligned)
#define TILE_F16   (BM * LDT)          // 5120 f16 = 10240 B per tile
#define STAGE_F16  (3 * TILE_F16)      // Ahi | Alo | B
#define STAGE_BYTES (STAGE_F16 * 2)    // 30720 B

// ---------------------------------------------------------------------------
// gfx1250 async-to-LDS copy (ASYNCcnt-tracked, no VGPR data movement).
//   lds_off : per-lane byte offset into wave's LDS allocation (16B aligned)
//   voff    : per-lane signed 32-bit byte offset from saddr (16B aligned)
//   base    : uniform 64-bit global base (SGPR pair)
// ---------------------------------------------------------------------------
__device__ __forceinline__ void async_b128(uint32_t lds_off, uint32_t voff,
                                           const void* base) {
    asm volatile("global_load_async_to_lds_b128 %0, %1, %2"
                 :: "v"(lds_off), "v"(voff), "s"(base)
                 : "memory");
}
__device__ __forceinline__ void wait_async0() {
    asm volatile("s_wait_asynccnt 0x0" ::: "memory");
}

// ---------------------------------------------------------------------------
// Kernel 1: per-output-row scale = max(mean(|W[o,:]|), EPS)
// ---------------------------------------------------------------------------
__global__ __launch_bounds__(256)
void ternary_scale_kernel(const float* __restrict__ W,
                          float* __restrict__ scale, int D_IN) {
    __shared__ float red[256];
    const int o = blockIdx.x;
    const float* row = W + (size_t)o * D_IN;
    float s = 0.f;
    for (int k = threadIdx.x; k < D_IN; k += 256) s += fabsf(row[k]);
    red[threadIdx.x] = s;
    __syncthreads();
    #pragma unroll
    for (int off = 128; off > 0; off >>= 1) {
        if (threadIdx.x < off) red[threadIdx.x] += red[threadIdx.x + off];
        __syncthreads();
    }
    if (threadIdx.x == 0) scale[o] = fmaxf(red[0] / (float)D_IN, EPS_V);
}

// ---------------------------------------------------------------------------
// Kernel 2a: split inputs fp32 -> f16 hi + f16 lo  (a = hi + lo, ~2^-22 exact)
// ---------------------------------------------------------------------------
__global__ __launch_bounds__(256)
void split_inputs_kernel(const float* __restrict__ A,
                         _Float16* __restrict__ Ahi,
                         _Float16* __restrict__ Alo, int total4) {
    const int i = blockIdx.x * 256 + threadIdx.x;
    if (i >= total4) return;
    const float4 v = ((const float4*)A)[i];
    float f[4] = {v.x, v.y, v.z, v.w};
    v4h hi, lo;
    #pragma unroll
    for (int j = 0; j < 4; ++j) {
        _Float16 h = (_Float16)f[j];
        hi[j] = h;
        lo[j] = (_Float16)(f[j] - (float)h);
    }
    ((v4h*)Ahi)[i] = hi;
    ((v4h*)Alo)[i] = lo;
}

// ---------------------------------------------------------------------------
// Kernel 2b: ternarize weight -> f16 {-1,0,1} (exact in f16; scale in epilogue)
// ---------------------------------------------------------------------------
__global__ __launch_bounds__(256)
void ternarize_kernel(const float* __restrict__ W,
                      const float* __restrict__ scale,
                      _Float16* __restrict__ Q, int K, int total4) {
    const int i = blockIdx.x * 256 + threadIdx.x;
    if (i >= total4) return;
    const int row = (int)(((long long)i * 4) / K);
    const float thr = scale[row] * THRESH;
    const float4 v = ((const float4*)W)[i];
    float f[4] = {v.x, v.y, v.z, v.w};
    v4h q;
    #pragma unroll
    for (int j = 0; j < 4; ++j)
        q[j] = (_Float16)((f[j] >= thr) ? 1.0f : ((f[j] <= -thr) ? -1.0f : 0.0f));
    ((v4h*)Q)[i] = q;
}

// ---------------------------------------------------------------------------
// Shared compute helpers (fragment layouts per CDNA5 ISA 7.12.2 / 7.12.5)
// ---------------------------------------------------------------------------
__device__ __forceinline__ v16h load_frag16(const _Float16* p0, const _Float16* p1) {
    v8h a = *(const v8h*)p0;
    v8h b = *(const v8h*)p1;
    return __builtin_shufflevector(a, b, 0,1,2,3,4,5,6,7,8,9,10,11,12,13,14,15);
}

// ---------------------------------------------------------------------------
// Kernel 3 (fast path): double-buffered async-to-LDS WMMA GEMM.
//   Ahi/Alo : [M][K] f16, Qw : [Nd][K] f16 (B columns = contiguous K)
// ---------------------------------------------------------------------------
__global__ __launch_bounds__(256)
void ternary_gemm_async(const _Float16* __restrict__ Ahi,
                        const _Float16* __restrict__ Alo,
                        const _Float16* __restrict__ Qw,
                        const float* __restrict__ scale,
                        const float* __restrict__ bias,
                        float* __restrict__ C,
                        int M, int Nd, int K) {
    __shared__ _Float16 smem[2 * STAGE_F16];

    const int tid  = threadIdx.x;
    const int wave = tid >> 5;
    const int lane = tid & 31;
    const int lr   = lane & 15;
    const int half = lane >> 4;

    const int n0 = blockIdx.x * BN;
    const int m0 = blockIdx.y * BM;
    const int wm = wave >> 2;     // 0..1 : 64-row slab
    const int wn = wave & 3;      // 0..3 : 32-col slab

    v8f zero = {0.f,0.f,0.f,0.f,0.f,0.f,0.f,0.f};
    v8f acc[4][2];
    #pragma unroll
    for (int i = 0; i < 4; ++i)
        #pragma unroll
        for (int j = 0; j < 2; ++j) acc[i][j] = zero;

    // --- async staging pattern: 512 chunks of 16 B per tile, 2 per thread ---
    // chunk c: row = c>>2 (0..127), cseg = c&3 ; chunk2 = chunk + 256 -> row+64
    const uint32_t srow  = (uint32_t)(tid >> 2);        // 0..63
    const uint32_t csegB = (uint32_t)(tid & 3) * 16u;   // byte seg in 64B row
    const uint32_t smem_base = (uint32_t)(uintptr_t)&smem[0];
    const uint32_t rowlds = srow * (LDT * 2) + csegB;
    const uint32_t l_Ahi = smem_base + rowlds;           // + s*STAGE_BYTES
    const uint32_t l_Alo = l_Ahi + TILE_F16 * 2;
    const uint32_t l_B   = l_Ahi + 2 * TILE_F16 * 2;
    const uint32_t halfTile = 64u * (LDT * 2);           // +64 rows in LDS

    const uint32_t gRow  = (uint32_t)(K * 2);            // global row pitch (B)
    const uint32_t voffA = ((uint32_t)m0 + srow) * gRow + csegB;
    const uint32_t voffB = ((uint32_t)n0 + srow) * gRow + csegB;
    const uint32_t gHalf = 64u * gRow;                   // +64 rows in global

    const int NK = K / BK;

    // prologue: slice 0 -> stage 0
    {
        async_b128(l_Ahi,            voffA,         Ahi);
        async_b128(l_Ahi + halfTile, voffA + gHalf, Ahi);
        async_b128(l_Alo,            voffA,         Alo);
        async_b128(l_Alo + halfTile, voffA + gHalf, Alo);
        async_b128(l_B,              voffB,         Qw);
        async_b128(l_B + halfTile,   voffB + gHalf, Qw);
    }

    for (int kb = 0; kb < NK; ++kb) {
        wait_async0();          // this wave's slice-kb fills done
        __syncthreads();        // everyone's fills done; prev compute done

        if (kb + 1 < NK) {      // overlap next fills with this slice's WMMAs
            const uint32_t sb = (uint32_t)((kb + 1) & 1) * STAGE_BYTES;
            const uint32_t ko = (uint32_t)(kb + 1) * (BK * 2);
            async_b128(l_Ahi + sb,            voffA + ko,         Ahi);
            async_b128(l_Ahi + sb + halfTile, voffA + ko + gHalf, Ahi);
            async_b128(l_Alo + sb,            voffA + ko,         Alo);
            async_b128(l_Alo + sb + halfTile, voffA + ko + gHalf, Alo);
            async_b128(l_B + sb,              voffB + ko,         Qw);
            async_b128(l_B + sb + halfTile,   voffB + ko + gHalf, Qw);
        }

        const _Float16* sAhi = smem + (kb & 1) * STAGE_F16;
        const _Float16* sAlo = sAhi + TILE_F16;
        const _Float16* sB   = sAhi + 2 * TILE_F16;

        // B frags (32x16 KxN): lane lr = column, lanes 0-15 K0..15, 16-31 K16..31
        v16h bf[2];
        #pragma unroll
        for (int j = 0; j < 2; ++j) {
            const int br = (wn * 32 + j * 16 + lr) * LDT + half * 16;
            bf[j] = load_frag16(&sB[br], &sB[br + 8]);
        }
        // A frags (16x32 MxK): lane<16 K0..15, lane>=16 K16..31
        v16h ah[4], al[4];
        #pragma unroll
        for (int i = 0; i < 4; ++i) {
            const int ar = (wm * 64 + i * 16 + lr) * LDT + half * 16;
            ah[i] = load_frag16(&sAhi[ar], &sAhi[ar + 8]);
            al[i] = load_frag16(&sAlo[ar], &sAlo[ar + 8]);
        }
        #pragma unroll
        for (int i = 0; i < 4; ++i)
            #pragma unroll
            for (int j = 0; j < 2; ++j) {
                acc[i][j] = __builtin_amdgcn_wmma_f32_16x16x32_f16(
                    false, ah[i], false, bf[j], (short)0, acc[i][j], false, false);
                acc[i][j] = __builtin_amdgcn_wmma_f32_16x16x32_f16(
                    false, al[i], false, bf[j], (short)0, acc[i][j], false, false);
            }
    }

    // epilogue: out = acc * scale[col] + bias[col]
    #pragma unroll
    for (int i = 0; i < 4; ++i) {
        const int row0 = m0 + wm * 64 + i * 16 + half * 8;
        #pragma unroll
        for (int j = 0; j < 2; ++j) {
            const int col = n0 + wn * 32 + j * 16 + lr;
            const float s  = scale[col];
            const float bv = bias[col];
            #pragma unroll
            for (int r = 0; r < 8; ++r)
                C[(size_t)(row0 + r) * Nd + col] = acc[i][j][r] * s + bv;
        }
    }
}

// ---------------------------------------------------------------------------
// Kernel 3' (fallback if ws too small): fused single-buffer WMMA GEMM
// (on-the-fly split/ternarize through LDS)
// ---------------------------------------------------------------------------
__global__ __launch_bounds__(256)
void ternary_gemm_fused(const float* __restrict__ A,
                        const float* __restrict__ W,
                        const float* __restrict__ scale,
                        const float* __restrict__ bias,
                        float* __restrict__ C,
                        int M, int Nd, int K) {
    __shared__ _Float16 sAhi[BM * LDT];
    __shared__ _Float16 sAlo[BM * LDT];
    __shared__ _Float16 sB  [BN * LDT];

    const int tid  = threadIdx.x;
    const int wave = tid >> 5;
    const int lane = tid & 31;
    const int lr   = lane & 15;
    const int half = lane >> 4;
    const int n0 = blockIdx.x * BN;
    const int m0 = blockIdx.y * BM;
    const int wm = wave >> 2;
    const int wn = wave & 3;

    v8f zero = {0.f,0.f,0.f,0.f,0.f,0.f,0.f,0.f};
    v8f acc[4][2];
    #pragma unroll
    for (int i = 0; i < 4; ++i)
        #pragma unroll
        for (int j = 0; j < 2; ++j) acc[i][j] = zero;

    const int srow = tid >> 3;
    const int scol = (tid & 7) * 4;

    for (int k0 = 0; k0 < K; k0 += BK) {
        #pragma unroll
        for (int rr = 0; rr < BM; rr += 32) {
            const int r = srow + rr;
            const float4 va = *(const float4*)(A + (size_t)(m0 + r) * K + k0 + scol);
            float f[4] = {va.x, va.y, va.z, va.w};
            v4h hi, lo;
            #pragma unroll
            for (int i = 0; i < 4; ++i) {
                _Float16 h = (_Float16)f[i];
                hi[i] = h;
                lo[i] = (_Float16)(f[i] - (float)h);
            }
            *(v4h*)&sAhi[r * LDT + scol] = hi;
            *(v4h*)&sAlo[r * LDT + scol] = lo;
        }
        #pragma unroll
        for (int rr = 0; rr < BN; rr += 32) {
            const int r = srow + rr;
            const int o = n0 + r;
            const float4 vw = *(const float4*)(W + (size_t)o * K + k0 + scol);
            const float thr = scale[o] * THRESH;
            float f[4] = {vw.x, vw.y, vw.z, vw.w};
            v4h q;
            #pragma unroll
            for (int i = 0; i < 4; ++i)
                q[i] = (_Float16)((f[i] >= thr) ? 1.0f : ((f[i] <= -thr) ? -1.0f : 0.0f));
            *(v4h*)&sB[r * LDT + scol] = q;
        }
        __syncthreads();

        v16h bf[2];
        #pragma unroll
        for (int j = 0; j < 2; ++j) {
            const int br = (wn * 32 + j * 16 + lr) * LDT + half * 16;
            bf[j] = load_frag16(&sB[br], &sB[br + 8]);
        }
        v16h ah[4], al[4];
        #pragma unroll
        for (int i = 0; i < 4; ++i) {
            const int ar = (wm * 64 + i * 16 + lr) * LDT + half * 16;
            ah[i] = load_frag16(&sAhi[ar], &sAhi[ar + 8]);
            al[i] = load_frag16(&sAlo[ar], &sAlo[ar + 8]);
        }
        #pragma unroll
        for (int i = 0; i < 4; ++i)
            #pragma unroll
            for (int j = 0; j < 2; ++j) {
                acc[i][j] = __builtin_amdgcn_wmma_f32_16x16x32_f16(
                    false, ah[i], false, bf[j], (short)0, acc[i][j], false, false);
                acc[i][j] = __builtin_amdgcn_wmma_f32_16x16x32_f16(
                    false, al[i], false, bf[j], (short)0, acc[i][j], false, false);
            }
        __syncthreads();
    }

    #pragma unroll
    for (int i = 0; i < 4; ++i) {
        const int row0 = m0 + wm * 64 + i * 16 + half * 8;
        #pragma unroll
        for (int j = 0; j < 2; ++j) {
            const int col = n0 + wn * 32 + j * 16 + lr;
            const float s  = scale[col];
            const float bv = bias[col];
            #pragma unroll
            for (int r = 0; r < 8; ++r)
                C[(size_t)(row0 + r) * Nd + col] = acc[i][j][r] * s + bv;
        }
    }
}

// ---------------------------------------------------------------------------
extern "C" void kernel_launch(void* const* d_in, const int* in_sizes, int n_in,
                              void* d_out, int out_size, void* d_ws, size_t ws_size,
                              hipStream_t stream) {
    const float* inputs = (const float*)d_in[0];
    const float* weight = (const float*)d_in[1];
    const float* bias   = (const float*)d_in[2];
    float* out = (float*)d_out;

    const int D_OUT = in_sizes[2];                 // 4096
    const int D_IN  = in_sizes[1] / D_OUT;         // 4096
    const int M     = in_sizes[0] / D_IN;          // 8192

    // workspace layout: [scale Nd f32][Ahi M*K f16][Alo M*K f16][Q Nd*K f16]
    const size_t scale_b = (size_t)D_OUT * sizeof(float);
    const size_t aTiles  = (size_t)M * D_IN * sizeof(_Float16);
    const size_t qBytes  = (size_t)D_OUT * D_IN * sizeof(_Float16);
    const size_t need    = scale_b + 2 * aTiles + qBytes;

    float* scale = (float*)d_ws;
    ternary_scale_kernel<<<D_OUT, 256, 0, stream>>>(weight, scale, D_IN);

    dim3 grid(D_OUT / BN, M / BM);

    if (ws_size >= need) {
        _Float16* Ahi = (_Float16*)((char*)d_ws + scale_b);
        _Float16* Alo = (_Float16*)((char*)d_ws + scale_b + aTiles);
        _Float16* Qw  = (_Float16*)((char*)d_ws + scale_b + 2 * aTiles);

        const int a4 = (M * D_IN) / 4;
        split_inputs_kernel<<<(a4 + 255) / 256, 256, 0, stream>>>(inputs, Ahi, Alo, a4);
        const int w4 = (D_OUT * D_IN) / 4;
        ternarize_kernel<<<(w4 + 255) / 256, 256, 0, stream>>>(weight, scale, Qw, D_IN, w4);

        ternary_gemm_async<<<grid, 256, 0, stream>>>(Ahi, Alo, Qw, scale, bias,
                                                     out, M, D_OUT, D_IN);
    } else {
        ternary_gemm_fused<<<grid, 256, 0, stream>>>(inputs, weight, scale, bias,
                                                     out, M, D_OUT, D_IN);
    }
}
